// HeteroGNN_14448269983897
// MI455X (gfx1250) — matrix operand order repaired and checked by reading, not compiled, not run on hardware
//
#include <hip/hip_runtime.h>
#include <hip/hip_bf16.h>

// ---------------------------------------------------------------------------
// MI455X (gfx1250) HeteroGNN forward.
// Dense math: v_wmma_f32_16x16x32_f16 (wave32 WMMA), f16 operands, f32 accum.
// GEMM uses register-prefetch pipelining: tile k+1 global loads are issued
// during tile k's WMMAs; uniform fast-path removes per-lane edge guards.
// Irregular math: f16 feature gathers (b128) + global f32 atomics for mean.
// ---------------------------------------------------------------------------

typedef __attribute__((ext_vector_type(16))) _Float16 v16h;
typedef __attribute__((ext_vector_type(8)))  _Float16 v8h;
typedef __attribute__((ext_vector_type(4)))  _Float16 v4h;
typedef __attribute__((ext_vector_type(8)))  float    v8f;
typedef __attribute__((ext_vector_type(4)))  float    v4f;

#define BM 128
#define BN 128
#define BK 32
#define LDS_PAD 8
#define LDS_STRIDE (BK + LDS_PAD)   // 40 halves = 80B row stride -> conflict-free b128 reads

// C[MxN] = (INIT ? 0 : C) + A[MxK] * Bt[NxK]^T + bias[n] ; optional ReLU.
// A: TA (float or _Float16), row-major MxK.  Bt: f16, row-major NxK (pre-transposed B).
// C: TC (float or _Float16), row-major MxN.
template <typename TA, typename TC, bool INIT, bool RELU>
__launch_bounds__(256)
__global__ void gemm_wmma_kernel(const TA* __restrict__ A,
                                 const _Float16* __restrict__ Bt,
                                 const float* __restrict__ bias,
                                 TC* __restrict__ C,
                                 int M, int N, int K)
{
    __shared__ __align__(16) _Float16 sA[BM * LDS_STRIDE];
    __shared__ __align__(16) _Float16 sB[BN * LDS_STRIDE];

    const int tid   = threadIdx.x;
    const int lane  = tid & 31;
    const int wave  = tid >> 5;          // 0..7
    const int waveM = wave >> 2;         // 0..1  -> 64 rows each
    const int waveN = wave & 3;          // 0..3  -> 32 cols each
    const int hlf   = lane >> 4;         // lane half (0/1)
    const int l16   = lane & 15;

    const int rowBase = blockIdx.y * BM;
    const int colBase = blockIdx.x * BN;
    const bool fullM  = (rowBase + BM) <= M;

    // ---- accumulator init: bias broadcast and/or existing C ----
    v8f acc[4][2];
#pragma unroll
    for (int mi = 0; mi < 4; ++mi) {
#pragma unroll
        for (int ni = 0; ni < 2; ++ni) {
            const int col = colBase + waveN * 32 + ni * 16 + l16;
            const float b = (bias != nullptr) ? bias[col] : 0.0f;
            if (INIT) {
#pragma unroll
                for (int v = 0; v < 8; ++v) acc[mi][ni][v] = b;
            } else {
                const int rb = rowBase + waveM * 64 + mi * 16 + hlf * 8;
                if (fullM) {
#pragma unroll
                    for (int v = 0; v < 8; ++v)
                        acc[mi][ni][v] = (float)C[(size_t)(rb + v) * N + col] + b;
                } else {
#pragma unroll
                    for (int v = 0; v < 8; ++v) {
                        const int r = rb + v;
                        float cv = (r < M) ? (float)C[(size_t)r * N + col] : 0.0f;
                        acc[mi][ni][v] = cv + b;
                    }
                }
            }
        }
    }

    // ---- tile prefetch registers ----
    v4f pa32[4];    // A prefetch (f32 path)
    v4h pa16[4];    // A prefetch (f16 path)
    v8h pb[2];      // B prefetch

    // fetch tile (k0) from global into registers
    auto fetch = [&](int k0) {
        const bool full = fullM && ((k0 + BK) <= K);
        if (full) {
#pragma unroll
            for (int it = 0; it < 4; ++it) {
                const int linear = tid + it * 256;           // 0..1023 4-elem slots
                const int gr = rowBase + (linear >> 3);
                const int gk = k0 + ((linear & 7) << 2);
                if (sizeof(TA) == 4)
                    pa32[it] = *(const v4f*)((const float*)A + (size_t)gr * K + gk);
                else
                    pa16[it] = *(const v4h*)((const _Float16*)A + (size_t)gr * K + gk);
            }
#pragma unroll
            for (int it = 0; it < 2; ++it) {
                const int linear = tid + it * 256;           // 0..511 8-elem slots
                const int gn = colBase + (linear >> 2);
                const int gk = k0 + ((linear & 3) << 3);
                pb[it] = *(const v8h*)(Bt + (size_t)gn * K + gk);
            }
        } else {
#pragma unroll
            for (int it = 0; it < 4; ++it) {
                const int linear = tid + it * 256;
                const int gr = rowBase + (linear >> 3);
                const int gk = k0 + ((linear & 7) << 2);
                const bool ok = (gr < M) && (gk < K);
                if (sizeof(TA) == 4) {
                    v4f v = {};
                    if (ok) v = *(const v4f*)((const float*)A + (size_t)gr * K + gk);
                    pa32[it] = v;
                } else {
                    v4h v = {};
                    if (ok) v = *(const v4h*)((const _Float16*)A + (size_t)gr * K + gk);
                    pa16[it] = v;
                }
            }
#pragma unroll
            for (int it = 0; it < 2; ++it) {
                const int linear = tid + it * 256;
                const int gn = colBase + (linear >> 2);
                const int gk = k0 + ((linear & 3) << 3);
                v8h v = {};
                if (gn < N && gk < K)
                    v = *(const v8h*)(Bt + (size_t)gn * K + gk);
                pb[it] = v;
            }
        }
    };

    // write prefetched tile into LDS (with f32->f16 convert on the A path)
    auto stage = [&]() {
#pragma unroll
        for (int it = 0; it < 4; ++it) {
            const int linear = tid + it * 256;
            const int row = linear >> 3;
            const int c4  = (linear & 7) << 2;
            v4h h;
            if (sizeof(TA) == 4) {
                h[0] = (_Float16)pa32[it][0]; h[1] = (_Float16)pa32[it][1];
                h[2] = (_Float16)pa32[it][2]; h[3] = (_Float16)pa32[it][3];
            } else {
                h = pa16[it];
            }
            *(v4h*)&sA[row * LDS_STRIDE + c4] = h;
        }
#pragma unroll
        for (int it = 0; it < 2; ++it) {
            const int linear = tid + it * 256;
            const int row = linear >> 2;
            const int c8  = (linear & 3) << 3;
            *(v8h*)&sB[row * LDS_STRIDE + c8] = pb[it];
        }
    };

    // ---- main K loop: fetch(k+1) overlaps wmma(k) ----
    fetch(0);
    for (int k0 = 0; k0 < K; k0 += BK) {
        __syncthreads();                 // previous iteration's LDS reads done
        stage();
        if (k0 + BK < K) fetch(k0 + BK); // issue next tile's global loads now
        __syncthreads();                 // staged tile visible

        // A fragments: 16-bit A layout (ISA 7.12.2): VGPR0-3 = K(half*8 .. +7),
        // VGPR4-7 = K(16+half*8 .. +7), row = lane%16.
        v16h af[4];
#pragma unroll
        for (int mi = 0; mi < 4; ++mi) {
            const _Float16* p = &sA[(waveM * 64 + mi * 16 + l16) * LDS_STRIDE];
            const v8h lo = *(const v8h*)(p + hlf * 8);
            const v8h hi = *(const v8h*)(p + 16 + hlf * 8);
#pragma unroll
            for (int i = 0; i < 8; ++i) { af[mi][i] = lo[i]; af[mi][i + 8] = hi[i]; }
        }
        // B fragments: same K order as A (K(half*8..+7), K(16+half*8..+7)), col = lane%16.
        v16h bf[2];
#pragma unroll
        for (int ni = 0; ni < 2; ++ni) {
            const _Float16* p = &sB[(waveN * 32 + ni * 16 + l16) * LDS_STRIDE];
            const v8h lo = *(const v8h*)(p + hlf * 8);
            const v8h hi = *(const v8h*)(p + 16 + hlf * 8);
#pragma unroll
            for (int i = 0; i < 8; ++i) { bf[ni][i] = lo[i]; bf[ni][i + 8] = hi[i]; }
        }
#pragma unroll
        for (int mi = 0; mi < 4; ++mi)
#pragma unroll
            for (int ni = 0; ni < 2; ++ni)
                acc[mi][ni] = __builtin_amdgcn_wmma_f32_16x16x32_f16(
                    false, af[mi], false, bf[ni], (short)0, acc[mi][ni], false, false);
    }

    // ---- store (C/D layout: VGPR v -> M = v + 8*half, N = lane%16) ----
#pragma unroll
    for (int mi = 0; mi < 4; ++mi) {
#pragma unroll
        for (int ni = 0; ni < 2; ++ni) {
            const int col = colBase + waveN * 32 + ni * 16 + l16;
            const int rb  = rowBase + waveM * 64 + mi * 16 + hlf * 8;
            if (fullM) {
#pragma unroll
                for (int v = 0; v < 8; ++v) {
                    float x = acc[mi][ni][v];
                    if (RELU) x = fmaxf(x, 0.0f);
                    C[(size_t)(rb + v) * N + col] = (TC)x;
                }
            } else {
#pragma unroll
                for (int v = 0; v < 8; ++v) {
                    const int r = rb + v;
                    if (r < M) {
                        float x = acc[mi][ni][v];
                        if (RELU) x = fmaxf(x, 0.0f);
                        C[(size_t)r * N + col] = (TC)x;
                    }
                }
            }
        }
    }
}

// ---------------- segment-mean aggregation ----------------

__global__ void zero_kernel(float* __restrict__ p, size_t n) {
    const size_t i = (size_t)blockIdx.x * blockDim.x + threadIdx.x;
    if (i < n) p[i] = 0.0f;
}

// one wave per edge; each lane loads 8 contiguous f16 (b128), f32 atomics into sums
__global__ void scatter_sum_kernel(const _Float16* __restrict__ X,
                                   const int* __restrict__ src,
                                   const int* __restrict__ dst,
                                   float* __restrict__ sum,
                                   float* __restrict__ cnt, int E)
{
    const int lane = threadIdx.x & 31;
    const int e = blockIdx.x * (blockDim.x >> 5) + (threadIdx.x >> 5);
    if (e >= E) return;
    const int s = src[e];
    const int d = dst[e];
    const v8h x = *(const v8h*)(X + (size_t)s * 256 + lane * 8);
    float* sd = sum + (size_t)d * 256 + lane * 8;
#pragma unroll
    for (int j = 0; j < 8; ++j)
        atomicAdd(&sd[j], (float)x[j]);
    if (lane == 0) atomicAdd(&cnt[d], 1.0f);
}

__global__ void mean_div_kernel(float* __restrict__ sum, const float* __restrict__ cnt, size_t n) {
    const size_t i = (size_t)blockIdx.x * blockDim.x + threadIdx.x;
    if (i >= n) return;
    sum[i] = sum[i] / fmaxf(cnt[i >> 8], 1.0f);   // H == 256
}

// ---------------- weight transpose (KxN f32 -> NxK f16) ----------------

__global__ void transpose_w_kernel(const float* __restrict__ W, _Float16* __restrict__ Wt,
                                   int K, int N)
{
    const size_t i = (size_t)blockIdx.x * blockDim.x + threadIdx.x;
    if (i >= (size_t)K * N) return;
    const int k = (int)(i / N);
    const int n = (int)(i % N);
    Wt[(size_t)n * K + k] = (_Float16)W[i];
}

// ---------------- graph pooling + heads ----------------

__global__ void init_last_kernel(int* __restrict__ last_idx, int G) {
    const int i = blockIdx.x * blockDim.x + threadIdx.x;
    if (i < G) last_idx[i] = 0;
}

__global__ void last_idx_kernel(const int* __restrict__ batch, int* __restrict__ last_idx,
                                int n, int G)
{
    const int i = blockIdx.x * blockDim.x + threadIdx.x;
    if (i >= n) return;
    const int g = batch[i];
    if ((i == n - 1 || batch[i + 1] != g) && g >= 0 && g < G) last_idx[g] = i;
}

__global__ void heads_kernel(const _Float16* __restrict__ feat_a,
                             const int* __restrict__ last_idx,
                             const float* __restrict__ Wa, const float* __restrict__ ba,
                             const float* __restrict__ Wtm, const float* __restrict__ btm,
                             const float* __restrict__ Wrm, const float* __restrict__ brm,
                             float* __restrict__ out, int Cn, int N_ACT)
{
    __shared__ float row[256];
    const int g = blockIdx.x;
    const int G = gridDim.x;
    int idx = last_idx[g];
    if (idx < 0) idx = 0;
    if (idx >= N_ACT) idx = N_ACT - 1;
    for (int k = threadIdx.x; k < 256; k += blockDim.x)
        row[k] = (float)feat_a[(size_t)idx * 256 + k];
    __syncthreads();
    const int j = threadIdx.x;
    if (j < Cn) {
        float a = ba[j];
        for (int k = 0; k < 256; ++k) a += row[k] * Wa[k * Cn + j];
        out[(size_t)g * Cn + j] = a;
    } else if (j == Cn) {
        float a = btm[0];
        for (int k = 0; k < 256; ++k) a += row[k] * Wtm[k];
        out[(size_t)G * Cn + g] = a;
    } else if (j == Cn + 1) {
        float a = brm[0];
        for (int k = 0; k < 256; ++k) a += row[k] * Wrm[k];
        out[(size_t)G * Cn + G + g] = a;
    }
}

// ---------------- host-side dispatch ----------------

static inline void launch_gemm(hipStream_t st, bool a_f16, bool c_f16, bool init, bool relu,
                               const void* A, const _Float16* Bt, const float* bias,
                               void* C, int M, int N, int K)
{
    dim3 grid((unsigned)((N + BN - 1) / BN), (unsigned)((M + BM - 1) / BM));
    dim3 block(256);
    if (!a_f16 && c_f16 && init && !relu)
        gemm_wmma_kernel<float, _Float16, true, false><<<grid, block, 0, st>>>(
            (const float*)A, Bt, bias, (_Float16*)C, M, N, K);
    else if (!a_f16 && !c_f16 && init && !relu)
        gemm_wmma_kernel<float, float, true, false><<<grid, block, 0, st>>>(
            (const float*)A, Bt, bias, (float*)C, M, N, K);
    else if (!a_f16 && !c_f16 && !init && !relu)
        gemm_wmma_kernel<float, float, false, false><<<grid, block, 0, st>>>(
            (const float*)A, Bt, bias, (float*)C, M, N, K);
    else if (a_f16 && !c_f16 && !init && !relu)
        gemm_wmma_kernel<_Float16, float, false, false><<<grid, block, 0, st>>>(
            (const _Float16*)A, Bt, bias, (float*)C, M, N, K);
    else if (!a_f16 && c_f16 && init && relu)
        gemm_wmma_kernel<float, _Float16, true, true><<<grid, block, 0, st>>>(
            (const float*)A, Bt, bias, (_Float16*)C, M, N, K);
}

extern "C" void kernel_launch(void* const* d_in, const int* in_sizes, int n_in,
                              void* d_out, int out_size, void* d_ws, size_t ws_size,
                              hipStream_t stream)
{
    (void)n_in; (void)ws_size;
    const float* x_act    = (const float*)d_in[0];
    const float* x_res    = (const float*)d_in[1];
    const float* x_time   = (const float*)d_in[2];
    const int* src_f  = (const int*)d_in[3];   const int* dst_f  = (const int*)d_in[4];
    const int* src_ar = (const int*)d_in[5];   const int* dst_ar = (const int*)d_in[6];
    const int* src_ra = (const int*)d_in[7];   const int* dst_ra = (const int*)d_in[8];
    const int* src_at = (const int*)d_in[9];   const int* dst_at = (const int*)d_in[10];
    const int* src_ta = (const int*)d_in[11];  const int* dst_ta = (const int*)d_in[12];
    const int* batch  = (const int*)d_in[13];
    // d_in[14] = num_graphs scalar (device); G derived from out_size instead.
    const float* Wp_act  = (const float*)d_in[15];
    const float* Wp_res  = (const float*)d_in[16];
    const float* Wp_time = (const float*)d_in[17];
    const float* Wl   = (const float*)d_in[18];
    const float* bl   = (const float*)d_in[19];
    const float* Wr   = (const float*)d_in[20];
    const float* Wlin = (const float*)d_in[21];
    const float* blin = (const float*)d_in[22];
    const float* W_out_act  = (const float*)d_in[23];
    const float* b_out_act  = (const float*)d_in[24];
    const float* W_out_time = (const float*)d_in[25];
    const float* b_out_time = (const float*)d_in[26];
    const float* W_out_rem  = (const float*)d_in[27];
    const float* b_out_rem  = (const float*)d_in[28];

    const int H  = 256;
    const size_t HH = (size_t)H * H;
    const int N_ACT  = in_sizes[0] / 64;
    const int N_RES  = in_sizes[1] / 32;
    const int N_TIME = in_sizes[2] / 16;
    const int E  = in_sizes[3];
    const int Cn = in_sizes[24];                 // 50
    const int Gg = out_size / (Cn + 2);          // 512

    // ---- workspace bump allocator (256B aligned) ----
    char* ws = (char*)d_ws;
    size_t off = 0;
    auto alloc = [&](size_t bytes) -> void* {
        off = (off + 255) & ~(size_t)255;
        void* p = ws + off;
        off += bytes;
        return p;
    };
    _Float16* feat_a = (_Float16*)alloc((size_t)N_ACT  * H * sizeof(_Float16));
    _Float16* feat_r = (_Float16*)alloc((size_t)N_RES  * H * sizeof(_Float16));
    _Float16* feat_t = (_Float16*)alloc((size_t)N_TIME * H * sizeof(_Float16));
    float* acc_a = (float*)alloc((size_t)N_ACT  * H * sizeof(float));
    float* acc_r = (float*)alloc((size_t)N_RES  * H * sizeof(float));
    float* acc_t = (float*)alloc((size_t)N_TIME * H * sizeof(float));
    float* sum   = (float*)alloc((size_t)N_ACT  * H * sizeof(float));   // max n_dst
    float* cnt   = (float*)alloc((size_t)N_ACT * sizeof(float));
    _Float16* wt_Wp_act  = (_Float16*)alloc((size_t)64 * H * sizeof(_Float16));
    _Float16* wt_Wp_res  = (_Float16*)alloc((size_t)32 * H * sizeof(_Float16));
    _Float16* wt_Wp_time = (_Float16*)alloc((size_t)16 * H * sizeof(_Float16));
    _Float16* wt_Wl   = (_Float16*)alloc(10 * HH * sizeof(_Float16));
    _Float16* wt_Wr   = (_Float16*)alloc(10 * HH * sizeof(_Float16));
    _Float16* wt_Wlin = (_Float16*)alloc(6 * HH * sizeof(_Float16));
    int* last_idx = (int*)alloc((size_t)Gg * sizeof(int));

    // ---- one-time weight transposes (f32 KxN -> f16 NxK) ----
    auto tpose = [&](const float* W, _Float16* Wt, int K, int N) {
        const size_t n = (size_t)K * N;
        transpose_w_kernel<<<(unsigned)((n + 255) / 256), 256, 0, stream>>>(W, Wt, K, N);
    };
    tpose(Wp_act,  wt_Wp_act,  64, H);
    tpose(Wp_res,  wt_Wp_res,  32, H);
    tpose(Wp_time, wt_Wp_time, 16, H);
    for (int i = 0; i < 10; ++i) tpose(Wl  + i * HH, wt_Wl  + i * HH, H, H);
    for (int i = 0; i < 10; ++i) tpose(Wr  + i * HH, wt_Wr  + i * HH, H, H);
    for (int i = 0; i < 6;  ++i) tpose(Wlin + i * HH, wt_Wlin + i * HH, H, H);

    // ---- input projections (f32 in, f16 features out) ----
    launch_gemm(stream, false, true, true, false, x_act,  wt_Wp_act,  nullptr, feat_a, N_ACT,  H, 64);
    launch_gemm(stream, false, true, true, false, x_res,  wt_Wp_res,  nullptr, feat_r, N_RES,  H, 32);
    launch_gemm(stream, false, true, true, false, x_time, wt_Wp_time, nullptr, feat_t, N_TIME, H, 16);

    // ---- segment mean helper ----
    auto sage_mean = [&](const _Float16* X, const int* s, const int* d, int n_dst) {
        const size_t n = (size_t)n_dst * H;
        zero_kernel<<<(unsigned)((n + 255) / 256), 256, 0, stream>>>(sum, n);
        zero_kernel<<<(unsigned)((n_dst + 255) / 256), 256, 0, stream>>>(cnt, (size_t)n_dst);
        scatter_sum_kernel<<<(unsigned)((E + 7) / 8), 256, 0, stream>>>(X, s, d, sum, cnt, E);
        mean_div_kernel<<<(unsigned)((n + 255) / 256), 256, 0, stream>>>(sum, cnt, n);
    };

    // ---- 2 hetero layers ----
    for (int l = 0; l < 2; ++l) {
        const size_t L5 = (size_t)l * 5;
        const size_t L3 = (size_t)l * 3;
        // dst = act : follows (rel 0), res->act (rel 2), time->act (rel 4)
        sage_mean(feat_a, src_f, dst_f, N_ACT);
        launch_gemm(stream, false, false, true,  false, sum,    wt_Wl + (L5 + 0) * HH, bl + (L5 + 0) * H, acc_a, N_ACT, H, H);
        launch_gemm(stream, true,  false, false, false, feat_a, wt_Wr + (L5 + 0) * HH, nullptr,           acc_a, N_ACT, H, H);
        sage_mean(feat_r, src_ra, dst_ra, N_ACT);
        launch_gemm(stream, false, false, false, false, sum,    wt_Wl + (L5 + 2) * HH, bl + (L5 + 2) * H, acc_a, N_ACT, H, H);
        launch_gemm(stream, true,  false, false, false, feat_a, wt_Wr + (L5 + 2) * HH, nullptr,           acc_a, N_ACT, H, H);
        sage_mean(feat_t, src_ta, dst_ta, N_ACT);
        launch_gemm(stream, false, false, false, false, sum,    wt_Wl + (L5 + 4) * HH, bl + (L5 + 4) * H, acc_a, N_ACT, H, H);
        launch_gemm(stream, true,  false, false, false, feat_a, wt_Wr + (L5 + 4) * HH, nullptr,           acc_a, N_ACT, H, H);
        // dst = res : act->res (rel 1)
        sage_mean(feat_a, src_ar, dst_ar, N_RES);
        launch_gemm(stream, false, false, true,  false, sum,    wt_Wl + (L5 + 1) * HH, bl + (L5 + 1) * H, acc_r, N_RES, H, H);
        launch_gemm(stream, true,  false, false, false, feat_r, wt_Wr + (L5 + 1) * HH, nullptr,           acc_r, N_RES, H, H);
        // dst = time : act->time (rel 3)
        sage_mean(feat_a, src_at, dst_at, N_TIME);
        launch_gemm(stream, false, false, true,  false, sum,    wt_Wl + (L5 + 3) * HH, bl + (L5 + 3) * H, acc_t, N_TIME, H, H);
        launch_gemm(stream, true,  false, false, false, feat_t, wt_Wr + (L5 + 3) * HH, nullptr,           acc_t, N_TIME, H, H);
        // per-type Linear + ReLU back into f16 features
        launch_gemm(stream, false, true, true, true, acc_a, wt_Wlin + (L3 + 0) * HH, blin + (L3 + 0) * H, feat_a, N_ACT,  H, H);
        launch_gemm(stream, false, true, true, true, acc_r, wt_Wlin + (L3 + 1) * HH, blin + (L3 + 1) * H, feat_r, N_RES,  H, H);
        launch_gemm(stream, false, true, true, true, acc_t, wt_Wlin + (L3 + 2) * HH, blin + (L3 + 2) * H, feat_t, N_TIME, H, H);
    }

    // ---- pooling (last activity node per graph) + output heads ----
    init_last_kernel<<<(unsigned)((Gg + 255) / 256), 256, 0, stream>>>(last_idx, Gg);
    last_idx_kernel<<<(unsigned)((N_ACT + 255) / 256), 256, 0, stream>>>(batch, last_idx, N_ACT, Gg);
    heads_kernel<<<(unsigned)Gg, 64, 0, stream>>>(feat_a, last_idx,
                                                  W_out_act, b_out_act,
                                                  W_out_time, b_out_time,
                                                  W_out_rem, b_out_rem,
                                                  (float*)d_out, Cn, N_ACT);
}
